// MaskedCausalAttention_34239479283804
// MI455X (gfx1250) — matrix operand order repaired and checked by reading, not compiled, not run on hardware
//
#include <hip/hip_runtime.h>

typedef _Float16 half_t;
typedef __attribute__((ext_vector_type(16))) _Float16 v16h;
typedef __attribute__((ext_vector_type(8)))  _Float16 v8h;
typedef __attribute__((ext_vector_type(8)))  float    v8f;

#define B_SZ   4
#define T_SEQ  2048
#define C_DIM  1024
#define NHEAD  16
#define HDIM   64
#define M_ROWS (B_SZ * T_SEQ)   // 8192

// ---------------------------------------------------------------------------
// WMMA helpers (layouts per CDNA5 ISA 7.12.2, wave32)
// ---------------------------------------------------------------------------
__device__ __forceinline__ v8f wmma16(v16h a, v16h b, v8f c) {
  // D = A(16x32 f16) * B(32x16 f16) + C(16x16 f32)
  return __builtin_amdgcn_wmma_f32_16x16x32_f16(false, a, false, b, (short)0, c,
                                                false, false);
}

// A fragment: matrix rows = M dim, K contiguous with row stride ld (halves).
// lane m (0-15): halves 0-7 = K k0..k0+7,  halves 8-15 = K k0+16..k0+23
// lane m+16   : halves 0-7 = K k0+8..+15, halves 8-15 = K k0+24..+31
__device__ __forceinline__ v16h load_a_frag(const half_t* __restrict__ base,
                                            int ld, int row0, int k0, int lane) {
  const int m  = lane & 15;
  const int hi = lane >> 4;
  const half_t* p = base + (size_t)(row0 + m) * ld + k0;
  union { v16h v; v8h h[2]; } u;
  u.h[0] = *(const v8h*)(p + hi * 8);
  u.h[1] = *(const v8h*)(p + 16 + hi * 8);
  return u.v;
}

// B fragment: B[k][n] supplied from memory holding Bt[n][k] (K contiguous,
// row stride ld).  lane n (0-15): halves 0-7 = K k0..k0+7, 8-15 = K k0+16..+23;
// lane n+16: halves 0-7 = K k0+8..+15, 8-15 = K k0+24..+31
__device__ __forceinline__ v16h load_b_frag(const half_t* __restrict__ base,
                                            int ld, int n0, int k0, int lane) {
  const int n  = lane & 15;
  const int hi = lane >> 4;
  const half_t* p = base + (size_t)(n0 + n) * ld + k0;
  union { v16h v; v8h h[2]; } u;
  u.h[0] = *(const v8h*)(p + hi * 8);
  u.h[1] = *(const v8h*)(p + 16 + hi * 8);
  return u.v;
}

// ---------------------------------------------------------------------------
// Kernel 0: f32 -> f16 convert
// ---------------------------------------------------------------------------
__global__ void cvt_f32_f16(const float* __restrict__ src,
                            half_t* __restrict__ dst, int n) {
  int i = blockIdx.x * blockDim.x + threadIdx.x;
  int stride = gridDim.x * blockDim.x;
  for (; i < n; i += stride) dst[i] = (half_t)src[i];
}

// ---------------------------------------------------------------------------
// Kernel 1: QKV projection.  y = x @ W.T + b
// grid = (M/128, C/64, 3), block = 128 (4 waves).
// Each wave computes a 32x64 tile: B fragments reused across two 16-row
// tiles -> 12 b128 loads per 8 WMMAs per K-step (vs 10:4 with 16-row tiles).
// Q,K stored [B, N, T, D] f16;  V stored transposed [B, N, D, T] f16
// ---------------------------------------------------------------------------
__global__ void __launch_bounds__(128)
qkv_gemm(const half_t* __restrict__ xh,
         const half_t* __restrict__ wq, const half_t* __restrict__ wk,
         const half_t* __restrict__ wv,
         const float* __restrict__ bq, const float* __restrict__ bk,
         const float* __restrict__ bv,
         half_t* __restrict__ Qh, half_t* __restrict__ Kh,
         half_t* __restrict__ VTh) {
  const int lane = threadIdx.x & 31;
  const int wave = threadIdx.x >> 5;
  const int m0 = blockIdx.x * 128 + wave * 32;   // 32 rows per wave
  const int n0 = blockIdx.y * 64;
  const int which = blockIdx.z;

  const half_t* W   = (which == 0) ? wq : (which == 1) ? wk : wv;
  const float*  bia = (which == 0) ? bq : (which == 1) ? bk : bv;

  v8f acc[2][4] = {};
  for (int k = 0; k < C_DIM; k += 32) {
    v16h a0 = load_a_frag(xh, C_DIM, m0,      k, lane);
    v16h a1 = load_a_frag(xh, C_DIM, m0 + 16, k, lane);
#pragma unroll
    for (int i = 0; i < 4; ++i) {
      v16h b = load_b_frag(W, C_DIM, n0 + i * 16, k, lane);
      acc[0][i] = wmma16(a0, b, acc[0][i]);
      acc[1][i] = wmma16(a1, b, acc[1][i]);
    }
  }

  const int nl = lane & 15, hi = lane >> 4;
#pragma unroll
  for (int i = 0; i < 4; ++i) {
    const int o  = n0 + i * 16 + nl;       // output channel
    const float bb = bia[o];
    const int nh = o >> 6;                 // head
    const int d  = o & 63;                 // dim within head
#pragma unroll
    for (int rt = 0; rt < 2; ++rt) {
#pragma unroll
      for (int r = 0; r < 8; ++r) {
        const int m = m0 + rt * 16 + hi * 8 + r;   // global row = b*T + t
        const int b = m >> 11;
        const int t = m & (T_SEQ - 1);
        const half_t hv = (half_t)(acc[rt][i][r] + bb);
        if (which == 0) {
          Qh[(((size_t)(b * NHEAD + nh)) * T_SEQ + t) * HDIM + d] = hv;
        } else if (which == 1) {
          Kh[(((size_t)(b * NHEAD + nh)) * T_SEQ + t) * HDIM + d] = hv;
        } else {
          VTh[(((size_t)(b * NHEAD + nh)) * HDIM + d) * T_SEQ + t] = hv;
        }
      }
    }
  }
}

// ---------------------------------------------------------------------------
// Kernel 2: flash-style causal attention with clamp(max=1e4).
// grid = (T/64, B*N), block = 128: wave w owns query rows t0 = blk*64+w*16.
// Writes attn (heads merged) as f16 [B, T, C].
// ---------------------------------------------------------------------------
__global__ void __launch_bounds__(128)
attn_kernel(const half_t* __restrict__ Qh, const half_t* __restrict__ Kh,
            const half_t* __restrict__ VTh, half_t* __restrict__ Ah) {
  __shared__ float  sS[4][16][33];                 // raw (masked) scores
  __shared__ __align__(16) half_t sP[4][16][40];   // exp probs, f16
  __shared__ float  sAl[4][16];                    // per-row alpha
  __shared__ float  sLi[4][16];                    // per-row 1/l

  const int lane = threadIdx.x & 31;
  const int wave = threadIdx.x >> 5;
  const int nl = lane & 15, hi = lane >> 4;
  const int bh   = blockIdx.y;                     // (b*NHEAD + h)
  const int tblk = blockIdx.x;
  const int t0   = tblk * 64 + wave * 16;
  const int b    = bh >> 4;
  const int h    = bh & 15;

  const half_t* qbase = Qh  + (size_t)bh * T_SEQ * HDIM;
  const half_t* kbase = Kh  + (size_t)bh * T_SEQ * HDIM;
  const half_t* vbase = VTh + (size_t)bh * HDIM * T_SEQ;

  const v16h qA0 = load_a_frag(qbase, HDIM, t0, 0, lane);
  const v16h qA1 = load_a_frag(qbase, HDIM, t0, 32, lane);

  float m_i = -3.0e38f;     // running row max (row = nl, duplicated hi halves)
  float l_i = 0.0f;         // running row sum
  v8f  o[4] = {};
  const float scale = 0.125f;   // 1/sqrt(64)
  const int jsEnd = tblk * 64 + 63;   // uniform across block -> barriers legal

  for (int js = 0; js <= jsEnd; js += 32) {
    // ---- scores S = q @ k^T for s in [js, js+32) ----
    v16h kB00 = load_b_frag(kbase, HDIM, js,      0,  lane);
    v16h kB01 = load_b_frag(kbase, HDIM, js,      32, lane);
    v16h kB10 = load_b_frag(kbase, HDIM, js + 16, 0,  lane);
    v16h kB11 = load_b_frag(kbase, HDIM, js + 16, 32, lane);
    v8f s0 = {}, s1 = {};
    s0 = wmma16(qA0, kB00, s0);  s0 = wmma16(qA1, kB01, s0);
    s1 = wmma16(qA0, kB10, s1);  s1 = wmma16(qA1, kB11, s1);

    // ---- scale + causal mask + clamp, stage to LDS (C layout) ----
#pragma unroll
    for (int r = 0; r < 8; ++r) {
      const int row = hi * 8 + r;
      const int t   = t0 + row;
      const int sc0 = js + nl, sc1 = js + 16 + nl;
      float v0 = s0[r] * scale;
      float v1 = s1[r] * scale;
      v0 = (sc0 <= t) ? fminf(v0, 1.0e4f) : -3.0e38f;
      v1 = (sc1 <= t) ? fminf(v1, 1.0e4f) : -3.0e38f;
      sS[wave][row][nl]      = v0;
      sS[wave][row][16 + nl] = v1;
    }
    __syncthreads();

    // ---- online softmax: lane pair (nl, nl+16) handles row nl ----
    {
      float vals[16];
      float pm = -3.0e38f;
#pragma unroll
      for (int j = 0; j < 16; ++j) {
        vals[j] = sS[wave][nl][hi * 16 + j];
        pm = fmaxf(pm, vals[j]);
      }
      const float om   = __shfl_xor(pm, 16, 32);
      const float mnew = fmaxf(m_i, fmaxf(pm, om));
      const float alpha = __expf(m_i - mnew);
      float psum = 0.0f;
#pragma unroll
      for (int j = 0; j < 16; ++j) {
        const float p = __expf(vals[j] - mnew);
        psum += p;
        sP[wave][nl][hi * 16 + j] = (half_t)p;
      }
      const float osum = __shfl_xor(psum, 16, 32);
      l_i = l_i * alpha + psum + osum;
      m_i = mnew;
      if (hi == 0) sAl[wave][nl] = alpha;
    }
    __syncthreads();

    // ---- rescale O and accumulate O += P @ V ----
    float al[8];
#pragma unroll
    for (int r = 0; r < 8; ++r) al[r] = sAl[wave][hi * 8 + r];
#pragma unroll
    for (int i = 0; i < 4; ++i)
#pragma unroll
      for (int r = 0; r < 8; ++r) o[i][r] *= al[r];

    union { v16h v; v8h h2[2]; } up;                  // P as A fragment (LDS)
    up.h2[0] = *(const v8h*)&sP[wave][nl][hi * 8];
    up.h2[1] = *(const v8h*)&sP[wave][nl][16 + hi * 8];
    const v16h pA = up.v;
#pragma unroll
    for (int i = 0; i < 4; ++i) {
      v16h vB = load_b_frag(vbase, T_SEQ, i * 16, js, lane);
      o[i] = wmma16(pA, vB, o[i]);
    }
    __syncthreads();
  }

  if (hi == 0) sLi[wave][nl] = 1.0f / l_i;
  __syncthreads();
  float li[8];
#pragma unroll
  for (int r = 0; r < 8; ++r) li[r] = sLi[wave][hi * 8 + r];

  const size_t obase = ((size_t)b * T_SEQ + t0) * C_DIM + h * HDIM;
#pragma unroll
  for (int i = 0; i < 4; ++i)
#pragma unroll
    for (int r = 0; r < 8; ++r)
      Ah[obase + (size_t)(hi * 8 + r) * C_DIM + i * 16 + nl] =
          (half_t)(o[i][r] * li[r]);
}

// ---------------------------------------------------------------------------
// Kernel 3: out = x + attn @ Wp.T + bp   (f32 output)
// grid = (M/128, C/64), block = 128; 32x64 tile per wave (B-frag reuse)
// ---------------------------------------------------------------------------
__global__ void __launch_bounds__(128)
out_proj(const half_t* __restrict__ Ah, const half_t* __restrict__ wp,
         const float* __restrict__ bp, const float* __restrict__ x,
         float* __restrict__ out) {
  const int lane = threadIdx.x & 31;
  const int wave = threadIdx.x >> 5;
  const int m0 = blockIdx.x * 128 + wave * 32;
  const int n0 = blockIdx.y * 64;

  v8f acc[2][4] = {};
  for (int k = 0; k < C_DIM; k += 32) {
    v16h a0 = load_a_frag(Ah, C_DIM, m0,      k, lane);
    v16h a1 = load_a_frag(Ah, C_DIM, m0 + 16, k, lane);
#pragma unroll
    for (int i = 0; i < 4; ++i) {
      v16h b = load_b_frag(wp, C_DIM, n0 + i * 16, k, lane);
      acc[0][i] = wmma16(a0, b, acc[0][i]);
      acc[1][i] = wmma16(a1, b, acc[1][i]);
    }
  }

  const int nl = lane & 15, hi = lane >> 4;
#pragma unroll
  for (int i = 0; i < 4; ++i) {
    const int o = n0 + i * 16 + nl;
    const float bb = bp[o];
#pragma unroll
    for (int rt = 0; rt < 2; ++rt) {
#pragma unroll
      for (int r = 0; r < 8; ++r) {
        const int m = m0 + rt * 16 + hi * 8 + r;
        const size_t idx = (size_t)m * C_DIM + o;
        out[idx] = x[idx] + acc[rt][i][r] + bb;
      }
    }
  }
}

// ---------------------------------------------------------------------------
extern "C" void kernel_launch(void* const* d_in, const int* in_sizes, int n_in,
                              void* d_out, int out_size, void* d_ws,
                              size_t ws_size, hipStream_t stream) {
  const float* x  = (const float*)d_in[0];
  const float* Wq = (const float*)d_in[1];
  const float* bq = (const float*)d_in[2];
  const float* Wk = (const float*)d_in[3];
  const float* bk = (const float*)d_in[4];
  const float* Wv = (const float*)d_in[5];
  const float* bv = (const float*)d_in[6];
  const float* Wp = (const float*)d_in[7];
  const float* bp = (const float*)d_in[8];
  float* outp = (float*)d_out;

  char* ws = (char*)d_ws;
  const size_t xBytes = (size_t)M_ROWS * C_DIM * sizeof(half_t);   // 16 MB
  const size_t wBytes = (size_t)C_DIM * C_DIM * sizeof(half_t);    //  2 MB
  half_t* xh  = (half_t*)ws;               ws += xBytes;
  half_t* wqh = (half_t*)ws;               ws += wBytes;
  half_t* wkh = (half_t*)ws;               ws += wBytes;
  half_t* wvh = (half_t*)ws;               ws += wBytes;
  half_t* wph = (half_t*)ws;               ws += wBytes;
  half_t* Qh  = (half_t*)ws;               ws += xBytes;
  half_t* Kh  = (half_t*)ws;               ws += xBytes;
  half_t* VTh = (half_t*)ws;               ws += xBytes;
  half_t* Ah  = (half_t*)ws;               ws += xBytes;

  const int nX = M_ROWS * C_DIM;     // 8,388,608
  const int nW = C_DIM * C_DIM;      // 1,048,576
  cvt_f32_f16<<<2048, 256, 0, stream>>>(x,  xh,  nX);
  cvt_f32_f16<<<512,  256, 0, stream>>>(Wq, wqh, nW);
  cvt_f32_f16<<<512,  256, 0, stream>>>(Wk, wkh, nW);
  cvt_f32_f16<<<512,  256, 0, stream>>>(Wv, wvh, nW);
  cvt_f32_f16<<<512,  256, 0, stream>>>(Wp, wph, nW);

  qkv_gemm<<<dim3(M_ROWS / 128, C_DIM / 64, 3), 128, 0, stream>>>(
      xh, wqh, wkh, wvh, bq, bk, bv, Qh, Kh, VTh);

  attn_kernel<<<dim3(T_SEQ / 64, B_SZ * NHEAD), 128, 0, stream>>>(Qh, Kh, VTh,
                                                                  Ah);

  out_proj<<<dim3(M_ROWS / 128, C_DIM / 64), 128, 0, stream>>>(Ah, wph, bp, x,
                                                               outp);
}
